// GatingNetwork_23356032155703
// MI455X (gfx1250) — matrix operand order repaired and checked by reading, not compiled, not run on hardware
//
#include <hip/hip_runtime.h>
#include <hip/hip_bf16.h>
#include <cstdint>
#include <cstddef>

// Problem constants (match reference)
#define NT 131072   // tokens
#define KD 2048     // input dim
#define HD 256      // hidden dim
#define ED 64       // experts
#define BM 128      // tokens per block
#define NKC (KD / 32)

typedef __attribute__((ext_vector_type(16))) _Float16 v16h;
typedef __attribute__((ext_vector_type(8)))  float    v8f;

// ---- workspace layout (bytes) ----
#define W1H_OFF 0                    // 2048*256 fp16 = 1,048,576 B (pre-swizzled B frags)
#define W2H_OFF 1048576              // 256*64  fp16 =    32,768 B (pre-swizzled B frags)
#define ACC_OFF 1081344              // float[128]: [0,64)=token counts, [64,128)=prob sums

// ---- dynamic LDS layout (bytes) ----
#define STG      24576               // per ping-pong buffer: A frags 8192 + B frags 16384
#define BA_OFF   0                   // A frag offset within a stage buffer
#define BB_OFF   8192                // B frag offset within a stage buffer
// stage1 ping-pong occupies [0, 49152); h buffer reuses [0, 67584) after stage1
#define HS       264                 // h row stride (fp16), padded
#define W2_OFF   67584               // W2 frags: 32768 B (ends 100352)
#define CNT_OFF  100352              // float[64]
#define RMAX_OFF 100608              // float[128]
#define RDEN_OFF 101120              // float[128]
#define SMEM_BYTES 101632
#define LSTR 68                      // logits row stride (floats); reuses h region

// CDNA5 async global->LDS path (guarded: falls back to register relay if absent)
// Probe-confirmed param 0 type: addrspace(1) pointer to 'int __attribute__((vector_size(16)))'
#if defined(__AMDGCN__) && \
    __has_builtin(__builtin_amdgcn_global_load_async_to_lds_b128) && \
    __has_builtin(__builtin_amdgcn_s_wait_asynccnt)
#define USE_ASYNC_LDS 1
typedef int v4i_vs __attribute__((vector_size(16)));
typedef __attribute__((address_space(1))) v4i_vs* as1_v4i_p;
typedef __attribute__((address_space(3))) v4i_vs* as3_v4i_p;
#else
#define USE_ASYNC_LDS 0
#endif

union V16U { v16h v; uint4 q[2]; };

// ---------------- prep: zero global accumulators ----------------
__global__ void init_accum_kernel(float* accum) {
    accum[threadIdx.x] = 0.0f;     // 128 threads
}

// ---------------- prep: W1 fp32 -> fp16, B-fragment swizzle ----------------
// dest layout: chunk c (K/32) -> nt (H/16) -> lane(32) -> idx(16)
// lane = (h%16) + 16*(kk>=16), idx = kk%16   (kk = k%32)
__global__ void conv_w1_kernel(const float* __restrict__ W1, _Float16* __restrict__ w1h) {
    int gid = blockIdx.x * 256 + threadIdx.x;          // < 524288
    int k    = gid >> 8;
    int hcol = gid & 255;
    int c  = k >> 5, kk = k & 31;
    int nt = hcol >> 4;
    int lane = (hcol & 15) + ((kk & 16) ? 16 : 0);
    int idx  = kk & 15;
    w1h[(((c * 16 + nt) * 32 + lane) << 4) + idx] = (_Float16)W1[gid];
}

// ---------------- prep: W2 fp32 -> fp16, B-fragment swizzle ----------------
__global__ void conv_w2_kernel(const float* __restrict__ W2, _Float16* __restrict__ w2h) {
    int gid = blockIdx.x * 256 + threadIdx.x;          // < 16384
    int kH = gid >> 6;
    int e  = gid & 63;
    int c  = kH >> 5, kk = kH & 31;
    int nt = e >> 4;
    int lane = (e & 15) + ((kk & 16) ? 16 : 0);
    int idx  = kk & 15;
    w2h[(((c * 4 + nt) * 32 + lane) << 4) + idx] = (_Float16)W2[gid];
}

// convert 16 fp32 -> 16 fp16 and store into A-fragment layout (two b128 stores)
__device__ __forceinline__ void cvt_store_xfrag(_Float16* a0, _Float16* a1,
                                                float4 f0, float4 f1,
                                                float4 f2, float4 f3) {
    float fl[16];
    *(float4*)&fl[0]  = f0; *(float4*)&fl[4]  = f1;
    *(float4*)&fl[8]  = f2; *(float4*)&fl[12] = f3;
    __align__(16) _Float16 hv[16];
    #pragma unroll
    for (int i = 0; i < 16; ++i) hv[i] = (_Float16)fl[i];
    *(uint4*)a0 = *(const uint4*)&hv[0];
    *(uint4*)a1 = *(const uint4*)&hv[8];
}

// ---------------- fused router kernel ----------------
__global__ __launch_bounds__(256)
void moe_router_kernel(const float*    __restrict__ x,
                       const _Float16* __restrict__ w1h,
                       const float*    __restrict__ b1,
                       const _Float16* __restrict__ w2h,
                       const float*    __restrict__ b2,
                       float*          __restrict__ outIdx,   // [NT,2] indices (as float)
                       float*          __restrict__ outGate,  // [NT,2] gates
                       float*          __restrict__ accum) {  // [0,64) cnt, [64,128) prob
    extern __shared__ char smem[];
    _Float16* hbuf = (_Float16*)(smem + 0);
    _Float16* w2F  = (_Float16*)(smem + W2_OFF);
    float* cntA = (float*)(smem + CNT_OFF);
    float* rmax = (float*)(smem + RMAX_OFF);
    float* rden = (float*)(smem + RDEN_OFF);
    float* lbuf = (float*)(smem + 0);

    const int t    = threadIdx.x;
    const int lane = t & 31;
    const int wv   = t >> 5;          // wave 0..7
    const int blk  = blockIdx.x;

    // persistent: copy pre-swizzled W2 fragments to LDS (256 thr * 128 B)
    {
        const uint4* src = (const uint4*)w2h + t * 8;
        uint4*       dst = (uint4*)w2F + t * 8;
        #pragma unroll
        for (int i = 0; i < 8; ++i) dst[i] = src[i];
    }
    if (t < ED) cntA[t] = 0.0f;

    // ---------------- stage 1: h = relu(x @ W1 + b1), double-buffered ----------------
    v8f acc[16] = {};

    const int r     = t >> 1;          // token row in block 0..127
    const int mt    = r >> 4;          // M-tile 0..7
    const int li    = r & 15;          // lane-in-tile
    const int idxb  = (t & 1) << 3;    // fragment idx base (0 or 8)
    const int halfk = (t & 1) << 4;    // k offset within chunk (0 or 16)
    const float* xp = x + (size_t)(blk * BM + r) * KD + halfk;
    const int aoff0 = (mt * 32 + li + halfk) * 16;        // k = halfk+0..7 (idx 0..7)
    const int aoff1 = (mt * 32 + li + halfk) * 16 + 8;    // k = halfk+8..15 (idx 8..15)

    // prologue: stage chunk 0 into buffer 0
    {
        _Float16* aN = (_Float16*)(smem + BA_OFF);
        _Float16* bN = (_Float16*)(smem + BB_OFF);
        const float4* xv = (const float4*)xp;
        cvt_store_xfrag(aN + aoff0, aN + aoff1, xv[0], xv[1], xv[2], xv[3]);
#if USE_ASYNC_LDS
        {
            _Float16* s = const_cast<_Float16*>(w1h) + t * 32;
            _Float16* d = bN + t * 32;
            __builtin_amdgcn_global_load_async_to_lds_b128((as1_v4i_p)s, (as3_v4i_p)d,  0, 0);
            __builtin_amdgcn_global_load_async_to_lds_b128((as1_v4i_p)s, (as3_v4i_p)d, 16, 0);
            __builtin_amdgcn_global_load_async_to_lds_b128((as1_v4i_p)s, (as3_v4i_p)d, 32, 0);
            __builtin_amdgcn_global_load_async_to_lds_b128((as1_v4i_p)s, (as3_v4i_p)d, 48, 0);
            __builtin_amdgcn_s_wait_asynccnt(0);
        }
#else
        {
            const uint4* src = (const uint4*)w1h + t * 4;
            uint4*       dst = (uint4*)(bN + t * 32);
            dst[0] = src[0]; dst[1] = src[1]; dst[2] = src[2]; dst[3] = src[3];
        }
#endif
    }
    __syncthreads();

    #pragma unroll 2
    for (int kc = 0; kc < NKC; ++kc) {
        char* cbuf = smem + (kc & 1) * STG;
        char* nbuf = smem + ((kc + 1) & 1) * STG;
        _Float16* aC = (_Float16*)(cbuf + BA_OFF);
        _Float16* bC = (_Float16*)(cbuf + BB_OFF);
        _Float16* aN = (_Float16*)(nbuf + BA_OFF);
        _Float16* bN = (_Float16*)(nbuf + BB_OFF);
        const bool haveNext = (kc + 1) < NKC;

        // issue next-chunk global traffic early so it overlaps the WMMA block
        float4 fx0, fx1, fx2, fx3;
        if (haveNext) {
            const float4* xv = (const float4*)(xp + (kc + 1) * 32);
            fx0 = xv[0]; fx1 = xv[1]; fx2 = xv[2]; fx3 = xv[3];
#if USE_ASYNC_LDS
            _Float16* s = const_cast<_Float16*>(w1h) + (size_t)(kc + 1) * 8192 + t * 32;
            _Float16* d = bN + t * 32;
            __builtin_amdgcn_global_load_async_to_lds_b128((as1_v4i_p)s, (as3_v4i_p)d,  0, 0);
            __builtin_amdgcn_global_load_async_to_lds_b128((as1_v4i_p)s, (as3_v4i_p)d, 16, 0);
            __builtin_amdgcn_global_load_async_to_lds_b128((as1_v4i_p)s, (as3_v4i_p)d, 32, 0);
            __builtin_amdgcn_global_load_async_to_lds_b128((as1_v4i_p)s, (as3_v4i_p)d, 48, 0);
#endif
        }
#if !USE_ASYNC_LDS
        uint4 w0, w1, w2, w3;
        if (haveNext) {
            const uint4* src = (const uint4*)(w1h + (size_t)(kc + 1) * 8192) + t * 4;
            w0 = src[0]; w1 = src[1]; w2 = src[2]; w3 = src[3];
        }
#endif

        // compute on current buffer
        V16U a;
        a.q[0] = *(const uint4*)(aC + (wv * 32 + lane) * 16);
        a.q[1] = *(const uint4*)(aC + (wv * 32 + lane) * 16 + 8);
        #pragma unroll
        for (int nt = 0; nt < 16; ++nt) {
            V16U b;
            b.q[0] = *(const uint4*)(bC + (nt * 32 + lane) * 16);
            b.q[1] = *(const uint4*)(bC + (nt * 32 + lane) * 16 + 8);
            acc[nt] = __builtin_amdgcn_wmma_f32_16x16x32_f16(
                false, a.v, false, b.v, (short)0, acc[nt], false, false);
        }

        // commit next buffer, then single barrier per iteration
        if (haveNext) {
            cvt_store_xfrag(aN + aoff0, aN + aoff1, fx0, fx1, fx2, fx3);
#if USE_ASYNC_LDS
            __builtin_amdgcn_s_wait_asynccnt(0);
#else
            uint4* dst = (uint4*)(bN + t * 32);
            dst[0] = w0; dst[1] = w1; dst[2] = w2; dst[3] = w3;
#endif
        }
        __syncthreads();
    }

    // bias + ReLU, write h (fp16, row-major padded) to LDS
    const int rowBase = wv * 16 + (lane >> 4) * 8;   // C-layout row base
    const int colLane = lane & 15;
    #pragma unroll
    for (int nt = 0; nt < 16; ++nt) {
        int col = nt * 16 + colLane;
        float bias = b1[col];
        #pragma unroll
        for (int rr = 0; rr < 8; ++rr) {
            float v = acc[nt][rr] + bias;
            hbuf[(rowBase + rr) * HS + col] = (_Float16)(v > 0.0f ? v : 0.0f);
        }
    }
    __syncthreads();

    // ---------------- stage 2: logits = h @ W2 + b2 ----------------
    v8f acc2[4] = {};
    {
        const int row = wv * 16 + colLane;
        #pragma unroll
        for (int kc = 0; kc < HD / 32; ++kc) {
            int kb = kc * 32 + ((lane < 16) ? 0 : 16);
            V16U a;
            a.q[0] = *(const uint4*)&hbuf[row * HS + kb];
            a.q[1] = *(const uint4*)&hbuf[row * HS + kb + 8];
            #pragma unroll
            for (int nt = 0; nt < 4; ++nt) {
                V16U b;
                b.q[0] = *(const uint4*)(w2F + ((kc * 4 + nt) * 32 + lane) * 16);
                b.q[1] = *(const uint4*)(w2F + ((kc * 4 + nt) * 32 + lane) * 16 + 8);
                acc2[nt] = __builtin_amdgcn_wmma_f32_16x16x32_f16(
                    false, a.v, false, b.v, (short)0, acc2[nt], false, false);
            }
        }
    }
    __syncthreads();   // all h reads done; safe to overwrite region with logits

    #pragma unroll
    for (int nt = 0; nt < 4; ++nt) {
        int e = nt * 16 + colLane;
        float bias = b2[e];
        #pragma unroll
        for (int rr = 0; rr < 8; ++rr)
            lbuf[(rowBase + rr) * LSTR + e] = acc2[nt][rr] + bias;
    }
    __syncthreads();

    // ---------------- per-token: top-2, gates, softmax stats ----------------
    if (t < BM) {
        const float* lr = lbuf + t * LSTR;
        float m = lr[0]; int i0 = 0;
        for (int e = 1; e < ED; ++e) { float v = lr[e]; if (v > m) { m = v; i0 = e; } }
        float m2 = -3.4e38f; int i1 = (i0 == 0) ? 1 : 0;
        for (int e = 0; e < ED; ++e) {
            if (e == i0) continue;
            float v = lr[e]; if (v > m2) { m2 = v; i1 = e; }
        }
        float den = 0.0f;
        for (int e = 0; e < ED; ++e) den += expf(lr[e] - m);
        float e1 = expf(m2 - m);
        float g0 = 1.0f / (1.0f + e1);           // softmax over {top1, top2}
        size_t tok = (size_t)blk * BM + t;
        outIdx [tok * 2]     = (float)i0;
        outIdx [tok * 2 + 1] = (float)i1;
        outGate[tok * 2]     = g0;
        outGate[tok * 2 + 1] = 1.0f - g0;
        rmax[t] = m;
        rden[t] = den;
        atomicAdd(&cntA[i0], 1.0f);
        atomicAdd(&cntA[i1], 1.0f);
    }
    __syncthreads();

    // ---------------- per-expert: column sums of router probs ----------------
    if (t < ED) {
        float s = 0.0f;
        for (int row = 0; row < BM; ++row)
            s += expf(lbuf[row * LSTR + t] - rmax[row]) / rden[row];
        atomicAdd(&accum[64 + t], s);        // sum of router probs for expert t
        atomicAdd(&accum[t], cntA[t]);       // token count for expert t
    }
}

// ---------------- final: balance loss scalar ----------------
__global__ void balance_loss_kernel(const float* __restrict__ accum, float* __restrict__ out) {
    __shared__ float red[64];
    int t = threadIdx.x;
    const float invN = 1.0f / (float)NT;
    red[t] = (float)ED * (accum[t] * invN) * (accum[64 + t] * invN);
    __syncthreads();
    for (int s = 32; s > 0; s >>= 1) {
        if (t < s) red[t] += red[t + s];
        __syncthreads();
    }
    if (t == 0) out[0] = red[0];
}

extern "C" void kernel_launch(void* const* d_in, const int* in_sizes, int n_in,
                              void* d_out, int out_size, void* d_ws, size_t ws_size,
                              hipStream_t stream) {
    (void)in_sizes; (void)n_in; (void)out_size; (void)ws_size;
    const float* x  = (const float*)d_in[0];
    const float* W1 = (const float*)d_in[1];
    const float* b1 = (const float*)d_in[2];
    const float* W2 = (const float*)d_in[3];
    const float* b2 = (const float*)d_in[4];

    _Float16* w1h   = (_Float16*)((char*)d_ws + W1H_OFF);
    _Float16* w2h   = (_Float16*)((char*)d_ws + W2H_OFF);
    float*    accum = (float*)((char*)d_ws + ACC_OFF);

    float* out     = (float*)d_out;
    float* outIdx  = out;                        // [NT,2]
    float* outGate = out + (size_t)2 * NT;       // [NT,2]
    float* outLoss = out + (size_t)4 * NT;       // scalar

    init_accum_kernel<<<1, 128, 0, stream>>>(accum);
    conv_w1_kernel<<<(KD * HD) / 256, 256, 0, stream>>>(W1, w1h);
    conv_w2_kernel<<<(HD * ED) / 256, 256, 0, stream>>>(W2, w2h);
    moe_router_kernel<<<NT / BM, 256, SMEM_BYTES, stream>>>(
        x, w1h, b1, w2h, b2, outIdx, outGate, accum);
    balance_loss_kernel<<<1, 64, 0, stream>>>(accum, outLoss);
}
